// EdgeGatedGraphConv_2156073582920
// MI455X (gfx1250) — compile-verified, matrix operand order checked by reading, not run on hardware
//
#include <hip/hip_runtime.h>
#include <hip/hip_bf16.h>
#include <math.h>

#define DF 64

typedef __attribute__((ext_vector_type(16))) __bf16 v16bf;
typedef __attribute__((ext_vector_type(8)))  float  v8f;

__device__ __forceinline__ __bf16 f2bf(float f) {
  return (__bf16)f;   // native v_cvt_pk_bf16_f32 on gfx1250
}

// Fragment K index pattern (ISA 7.12.2, 16-bit A/B 16x32):
// lanes 0-15 hold K {0..7,16..23}, lanes 16-31 hold K {8..15,24..31}.
// Per lane, a fragment is two contiguous 8-float blocks -> 4x float4 loads.

// Build A fragments for a 16x64 row tile (two 16x32 frags) from row pointer.
__device__ __forceinline__ void load_afrags(const float* __restrict__ rowp,
                                            int hi, v16bf& a0, v16bf& a1) {
  const float4* rp4 = (const float4*)rowp;
  const int b = hi * 2;                 // float4 index of K-half base
  float4 q0 = rp4[b + 0],  q1 = rp4[b + 1];    // k = kh .. kh+7
  float4 q2 = rp4[b + 4],  q3 = rp4[b + 5];    // k = 16+kh .. 16+kh+7
  float4 q4 = rp4[b + 8],  q5 = rp4[b + 9];    // k = 32+kh ..
  float4 q6 = rp4[b + 12], q7 = rp4[b + 13];   // k = 48+kh ..
  a0[0]=f2bf(q0.x); a0[1]=f2bf(q0.y); a0[2]=f2bf(q0.z); a0[3]=f2bf(q0.w);
  a0[4]=f2bf(q1.x); a0[5]=f2bf(q1.y); a0[6]=f2bf(q1.z); a0[7]=f2bf(q1.w);
  a0[8]=f2bf(q2.x); a0[9]=f2bf(q2.y); a0[10]=f2bf(q2.z); a0[11]=f2bf(q2.w);
  a0[12]=f2bf(q3.x); a0[13]=f2bf(q3.y); a0[14]=f2bf(q3.z); a0[15]=f2bf(q3.w);
  a1[0]=f2bf(q4.x); a1[1]=f2bf(q4.y); a1[2]=f2bf(q4.z); a1[3]=f2bf(q4.w);
  a1[4]=f2bf(q5.x); a1[5]=f2bf(q5.y); a1[6]=f2bf(q5.z); a1[7]=f2bf(q5.w);
  a1[8]=f2bf(q6.x); a1[9]=f2bf(q6.y); a1[10]=f2bf(q6.z); a1[11]=f2bf(q6.w);
  a1[12]=f2bf(q7.x); a1[13]=f2bf(q7.y); a1[14]=f2bf(q7.z); a1[15]=f2bf(q7.w);
}

__device__ __forceinline__ int frag_k(int i, int lane) {
  int j = i >> 1, h = i & 1;
  return ((j & 4) ? 16 : 0) + ((j & 3) << 1) + h + ((lane >> 4) << 3);
}

// B fragment for 32x16 tile of row-major 64x64 weight at (k0, n0): column
// access, scalar loads (tiny, cache-resident, hoisted outside hot loops).
__device__ __forceinline__ v16bf load_bfrag(const float* __restrict__ W,
                                            int k0, int n0, int lane) {
  int n = lane & 15;
  v16bf b;
#pragma unroll
  for (int i = 0; i < 16; ++i) {
    int k = k0 + frag_k(i, lane);
    b[i] = f2bf(W[(size_t)k * DF + n0 + n]);
  }
  return b;
}

// ---------------------------------------------------------------------------
__global__ void __launch_bounds__(256)
zero_kernel(float* __restrict__ p, size_t n) {
  size_t i = (size_t)blockIdx.x * blockDim.x + threadIdx.x;
  size_t st = (size_t)gridDim.x * blockDim.x;
  for (; i < n; i += st) p[i] = 0.0f;
}

// ---------------------------------------------------------------------------
// Node-side GEMMs: e_src, e_dst, Bh, Ax  ([N,64] @ [64,64] + bias) via WMMA.
// The matrix loop is fully unrolled: each iteration uses a kernel-arg pointer
// directly (global stores, no FLAT), and its 8 B-fragments stay in registers.
// node_feats (25.6 MB, L2-resident) is re-streamed per matrix: free at 192 MB.
// ---------------------------------------------------------------------------
__global__ void __launch_bounds__(256)
node_gemm_kernel(const float* __restrict__ node,
                 const float* __restrict__ W0, const float* __restrict__ b0,
                 const float* __restrict__ W1, const float* __restrict__ b1,
                 const float* __restrict__ W2, const float* __restrict__ b2,
                 const float* __restrict__ W3, const float* __restrict__ b3,
                 float* __restrict__ o0, float* __restrict__ o1,
                 float* __restrict__ o2, float* __restrict__ o3, int N)
{
  const int lane = threadIdx.x & 31;
  const int wave = threadIdx.x >> 5;
  const int lo = lane & 15, hi = lane >> 4;
  const float* Ws[4] = {W0, W1, W2, W3};
  const float* bs[4] = {b0, b1, b2, b3};
  float*       os[4] = {o0, o1, o2, o3};

  const int tiles = (N + 15) >> 4;
  const int tstride = gridDim.x * 8;

#pragma unroll
  for (int w = 0; w < 4; ++w) {
    const float* W = Ws[w];
    float* out = os[w];
    v16bf Bf[8];
    float bcol[4];
#pragma unroll
    for (int t = 0; t < 4; ++t) {
      Bf[t * 2]     = load_bfrag(W, 0,  t * 16, lane);
      Bf[t * 2 + 1] = load_bfrag(W, 32, t * 16, lane);
      bcol[t] = bs[w][t * 16 + lo];
    }
    for (int tile = blockIdx.x * 8 + wave; tile < tiles; tile += tstride) {
      const int row0 = tile << 4;
      int ar = row0 + lo; if (ar >= N) ar = N - 1;      // clamp, keep EXEC full
      v16bf a0, a1;
      load_afrags(node + (size_t)ar * DF, hi, a0, a1);
      const bool full = (row0 + 16) <= N;
#pragma unroll
      for (int t = 0; t < 4; ++t) {
        v8f c = {};
        c = __builtin_amdgcn_wmma_f32_16x16x32_bf16(false, a0, false, Bf[t*2],
                                                    (short)0, c, false, false);
        c = __builtin_amdgcn_wmma_f32_16x16x32_bf16(false, a1, false, Bf[t*2+1],
                                                    (short)0, c, false, false);
        // one base address, 8 immediate-offset stores
        float* op = out + (size_t)(row0 + hi * 8) * DF + (t * 16 + lo);
        if (full) {
#pragma unroll
          for (int r = 0; r < 8; ++r) op[r * DF] = c[r] + bcol[t];
        } else {
#pragma unroll
          for (int r = 0; r < 8; ++r)
            if (row0 + hi * 8 + r < N) op[r * DF] = c[r] + bcol[t];
        }
      }
    }
  }
}

// ---------------------------------------------------------------------------
// Fused edge kernel: eg = edge_feats @ W_edge_gate + b (WMMA); then per row:
// m = eg + e_src[src] + e_dst[dst]; sigma = sigmoid(m); NT-store m; atomic
// accumulate Bh[src]*sigma and sigma at dst. Half-wave gathers are 64B
// contiguous and hit the L2-resident tables; m is streamed with TH=NT so the
// 410 MB stream does not evict the ~154 MB hot set from the 192 MB L2.
// ---------------------------------------------------------------------------
__global__ void __launch_bounds__(256)
edge_fused_kernel(const float* __restrict__ edge, const float* __restrict__ W,
                  const float* __restrict__ bias, const int* __restrict__ src,
                  const int* __restrict__ dst, const float* __restrict__ esrc,
                  const float* __restrict__ edst, const float* __restrict__ Bh,
                  float* __restrict__ m_out, float* __restrict__ sumh,
                  float* __restrict__ sums, int E)
{
  const int lane = threadIdx.x & 31;
  const int wave = threadIdx.x >> 5;
  const int lo = lane & 15, hi = lane >> 4;

  v16bf Bf[8];
  float bcol[4];
#pragma unroll
  for (int t = 0; t < 4; ++t) {
    Bf[t * 2]     = load_bfrag(W, 0,  t * 16, lane);
    Bf[t * 2 + 1] = load_bfrag(W, 32, t * 16, lane);
    bcol[t] = bias[t * 16 + lo];
  }

  const int tiles = (E + 15) >> 4;
  const int tstride = gridDim.x * 8;
  for (int tile = blockIdx.x * 8 + wave; tile < tiles; tile += tstride) {
    const int row0 = tile << 4;
    int ar = row0 + lo; if (ar >= E) ar = E - 1;
    const float* rowp = edge + (size_t)ar * DF;
    __builtin_prefetch(rowp + (size_t)tstride * 16 * DF, 0, 0); // global_prefetch_b8
    v16bf a0, a1;
    load_afrags(rowp, hi, a0, a1);

    v8f c[4];
#pragma unroll
    for (int t = 0; t < 4; ++t) {
      v8f acc = {};
      acc = __builtin_amdgcn_wmma_f32_16x16x32_bf16(false, a0, false, Bf[t*2],
                                                    (short)0, acc, false, false);
      acc = __builtin_amdgcn_wmma_f32_16x16x32_bf16(false, a1, false, Bf[t*2+1],
                                                    (short)0, acc, false, false);
      c[t] = acc;
    }

    const bool full = (row0 + 16) <= E;
#pragma unroll
    for (int r = 0; r < 8; ++r) {
      const int row = row0 + r + hi * 8;
      if (full || row < E) {
        const int s = src[row];
        const int d = dst[row];
        const float* es = esrc + (size_t)s * DF + lo;
        const float* ed = edst + (size_t)d * DF + lo;
        const float* bh = Bh   + (size_t)s * DF + lo;
        float* sh = sumh + (size_t)d * DF + lo;
        float* ss = sums + (size_t)d * DF + lo;
        float* mo = m_out + (size_t)row * DF + lo;
#pragma unroll
        for (int t = 0; t < 4; ++t) {
          float mv = c[t][r] + bcol[t] + es[t * 16] + ed[t * 16];
          float sg = 1.0f / (1.0f + __expf(-mv));
          __builtin_nontemporal_store(mv, mo + t * 16);
          atomicAdd(sh + t * 16, bh[t * 16] * sg);
          atomicAdd(ss + t * 16, sg);
        }
      }
    }
  }
}

// ---------------------------------------------------------------------------
// Node combine: x_pre = Ax + sumh/(sums+eps), fused per-column BN partials.
// ---------------------------------------------------------------------------
__global__ void __launch_bounds__(256)
node_combine_kernel(const float* __restrict__ Ax, const float* __restrict__ sumh,
                    const float* __restrict__ sums, float* __restrict__ x_pre,
                    float* __restrict__ s1g, float* __restrict__ s2g, size_t total)
{
  __shared__ float s1[DF], s2[DF];
  if (threadIdx.x < DF) { s1[threadIdx.x] = 0.0f; s2[threadIdx.x] = 0.0f; }
  __syncthreads();
  const int c = threadIdx.x & (DF - 1);   // stride is a multiple of 64
  float ls1 = 0.0f, ls2 = 0.0f;
  size_t st = (size_t)gridDim.x * blockDim.x;
  for (size_t i = (size_t)blockIdx.x * blockDim.x + threadIdx.x; i < total; i += st) {
    float xp = Ax[i] + sumh[i] / (sums[i] + 1e-6f);
    x_pre[i] = xp;
    ls1 += xp; ls2 += xp * xp;
  }
  atomicAdd(&s1[c], ls1);
  atomicAdd(&s2[c], ls2);
  __syncthreads();
  if (threadIdx.x < DF) {
    atomicAdd(&s1g[threadIdx.x], s1[threadIdx.x]);
    atomicAdd(&s2g[threadIdx.x], s2[threadIdx.x]);
  }
}

// Per-column sum / sum-of-squares over a [rows,64] array (edge BN stats).
// NT loads: pure stream, keep L2 for the node-side tables.
__global__ void __launch_bounds__(256)
colstats_kernel(const float* __restrict__ x, float* __restrict__ s1g,
                float* __restrict__ s2g, size_t total)
{
  __shared__ float s1[DF], s2[DF];
  if (threadIdx.x < DF) { s1[threadIdx.x] = 0.0f; s2[threadIdx.x] = 0.0f; }
  __syncthreads();
  const int c = threadIdx.x & (DF - 1);
  float ls1 = 0.0f, ls2 = 0.0f;
  size_t st = (size_t)gridDim.x * blockDim.x;
  for (size_t i = (size_t)blockIdx.x * blockDim.x + threadIdx.x; i < total; i += st) {
    float v = __builtin_nontemporal_load(x + i);
    ls1 += v; ls2 += v * v;
  }
  atomicAdd(&s1[c], ls1);
  atomicAdd(&s2[c], ls2);
  __syncthreads();
  if (threadIdx.x < DF) {
    atomicAdd(&s1g[threadIdx.x], s1[threadIdx.x]);
    atomicAdd(&s2g[threadIdx.x], s2[threadIdx.x]);
  }
}

// BatchNorm (biased var) -> SiLU -> residual add. Streams in, streams out: NT.
__global__ void __launch_bounds__(256)
bn_silu_res_kernel(const float* __restrict__ x, const float* __restrict__ resid,
                   const float* __restrict__ s1g, const float* __restrict__ s2g,
                   const float* __restrict__ gamma, const float* __restrict__ beta,
                   float* __restrict__ out, size_t total, float inv_n)
{
  size_t st = (size_t)gridDim.x * blockDim.x;
  for (size_t i = (size_t)blockIdx.x * blockDim.x + threadIdx.x; i < total; i += st) {
    int c = (int)(i & (DF - 1));
    float mu  = s1g[c] * inv_n;
    float var = s2g[c] * inv_n - mu * mu;
    float xv  = __builtin_nontemporal_load(x + i);
    float rv  = __builtin_nontemporal_load(resid + i);
    float xh  = (xv - mu) * rsqrtf(var + 1e-5f) * gamma[c] + beta[c];
    float sl  = xh / (1.0f + __expf(-xh));
    __builtin_nontemporal_store(rv + sl, out + i);
  }
}

// ---------------------------------------------------------------------------
extern "C" void kernel_launch(void* const* d_in, const int* in_sizes, int n_in,
                              void* d_out, int out_size, void* d_ws, size_t ws_size,
                              hipStream_t stream) {
  const float* node = (const float*)d_in[0];
  const float* edge = (const float*)d_in[1];
  const int*   src  = (const int*)d_in[2];
  const int*   dst  = (const int*)d_in[3];
  const float* Wsg = (const float*)d_in[4];  const float* bsg = (const float*)d_in[5];
  const float* Wdg = (const float*)d_in[6];  const float* bdg = (const float*)d_in[7];
  const float* Weg = (const float*)d_in[8];  const float* beg = (const float*)d_in[9];
  const float* Wsu = (const float*)d_in[10]; const float* bsu = (const float*)d_in[11];
  const float* Wdu = (const float*)d_in[12]; const float* bdu = (const float*)d_in[13];
  const float* gn  = (const float*)d_in[14]; const float* bn  = (const float*)d_in[15];
  const float* ge  = (const float*)d_in[16]; const float* be  = (const float*)d_in[17];

  const size_t N64 = (size_t)in_sizes[0];            // N*64
  const size_t E64 = (size_t)in_sizes[1];            // E*64
  const int N = (int)(N64 / DF);
  const int E = (int)(E64 / DF);

  float* w = (float*)d_ws;
  float* e_src = w;                    // [N,64]
  float* e_dst = e_src + N64;          // [N,64]
  float* Bh    = e_dst + N64;          // [N,64]
  float* Ax    = Bh    + N64;          // [N,64]
  float* sumh  = Ax    + N64;          // [N,64]
  float* sums  = sumh  + N64;          // [N,64]
  float* x_pre = sums  + N64;          // [N,64]
  float* stats = x_pre + N64;          // 4*64 floats
  float* nst1 = stats, *nst2 = stats + 64, *est1 = stats + 128, *est2 = stats + 192;
  float* m     = stats + 256;          // [E,64]

  float* xout = (float*)d_out;         // x: [N,64]
  float* yout = xout + N64;            // y: [E,64]

  // 1) zero accumulators
  zero_kernel<<<512, 256, 0, stream>>>(sumh, 2 * N64);
  zero_kernel<<<1, 256, 0, stream>>>(stats, 256);

  // 2) node GEMMs: e_src(Wsg), e_dst(Wdg), Bh(Wdu), Ax(Wsu)
  node_gemm_kernel<<<512, 256, 0, stream>>>(node, Wsg, bsg, Wdg, bdg,
                                            Wdu, bdu, Wsu, bsu,
                                            e_src, e_dst, Bh, Ax, N);

  // 3) fused edge GEMM + gather + sigmoid + segment-sum atomics + m store
  edge_fused_kernel<<<1024, 256, 0, stream>>>(edge, Weg, beg, src, dst,
                                              e_src, e_dst, Bh,
                                              m, sumh, sums, E);

  // 4) node combine + node BN partial stats
  node_combine_kernel<<<512, 256, 0, stream>>>(Ax, sumh, sums, x_pre,
                                               nst1, nst2, N64);

  // 5) edge BN partial stats over m
  colstats_kernel<<<1024, 256, 0, stream>>>(m, est1, est2, E64);

  // 6) epilogues: BN -> SiLU -> residual
  bn_silu_res_kernel<<<512, 256, 0, stream>>>(x_pre, node, nst1, nst2, gn, bn,
                                              xout, N64, 1.0f / (float)N);
  bn_silu_res_kernel<<<2048, 256, 0, stream>>>(m, edge, est1, est2, ge, be,
                                               yout, E64, 1.0f / (float)E);
}
